// RAN_79620103733393
// MI455X (gfx1250) — matrix operand order repaired and checked by reading, not compiled, not run on hardware
//
#include <hip/hip_runtime.h>

// ---- CDNA5 WMMA types (device-side only usage) ----
typedef __attribute__((ext_vector_type(16))) __bf16 v16bf;
typedef __attribute__((ext_vector_type(8)))  __bf16 v8bf;
typedef __attribute__((ext_vector_type(8)))  float  v8f;

#define INPUT_SZ 256
#define HIDDEN   512
#define GATES    1536   // 3*HIDDEN
#define HIST     2048
#define SEQ      512
#define BATCH    256
#define BT       16     // batch tile per workgroup

// ---------------------------------------------------------------------------
// Fragment loaders for CDNA5 16x16x32 bf16 WMMA, using this code's contiguous k (16h+s) in A and B
// (CDNA5 ISA §7.12.2).
// A (16xK, 16-bit): lane m<16 holds row m, K = k0+[0..15] contiguous;
// lanes 16-31 hold K offset by +16 (matches B order).
// B (Kx16): lane holds column N=lane%16, 16 contiguous K values starting at
// k0 + (lane/16)*16  -> contiguous 32B per lane when B-col = a weight row.
// C/D (16x16 f32): vgpr r of lane l = element (M = r + (l/16)*8, N = l%16).
// ---------------------------------------------------------------------------
static __device__ __forceinline__ v16bf load_a_frag(const __bf16* base, int ld,
                                                    int m, int half, int k0) {
  const __bf16* p = base + m * ld + k0 + half * 16;
  v8bf lo = *(const v8bf*)p;
  v8bf hi = *(const v8bf*)(p + 8);
  v16bf r;
#pragma unroll
  for (int i = 0; i < 8; ++i) { r[i] = lo[i]; r[i + 8] = hi[i]; }
  return r;
}

static __device__ __forceinline__ v16bf load_b_frag(const __bf16* w, int ld,
                                                    int j0, int k0, int lane) {
  const int n    = lane & 15;
  const int koff = (lane >> 4) << 4;
  return *(const v16bf*)(w + (size_t)(j0 + n) * ld + (k0 + koff));
}

static __device__ __forceinline__ v8f wmma_bf16(v16bf a, v16bf b, v8f c) {
  // (neg_a, A, neg_b, B, c_mod, C, reuse_a, reuse_b)
  return __builtin_amdgcn_wmma_f32_16x16x32_bf16(false, a, false, b, (short)0,
                                                 c, false, false);
}

static __device__ __forceinline__ float sigmoidf_(float v) {
  return 1.0f / (1.0f + __expf(-v));
}

static __device__ __forceinline__ unsigned short f2bf_bits(float f) {
  unsigned int u = __float_as_uint(f);
  unsigned int r = u + 0x7fffu + ((u >> 16) & 1u);   // round-to-nearest-even
  return (unsigned short)(r >> 16);
}

// ---------------------------------------------------------------------------
// Kernel 0: fp32 -> bf16 weight conversion (one-time; weights stay in L2)
// ---------------------------------------------------------------------------
__global__ void cvt_bf16_kernel(const float* __restrict__ src,
                                unsigned short* __restrict__ dst, int n) {
  int i = blockIdx.x * blockDim.x + threadIdx.x;
  const int stride = gridDim.x * blockDim.x;
  for (; i < n; i += stride) dst[i] = f2bf_bits(src[i]);
}

// ---------------------------------------------------------------------------
// Kernel 1: GRU recurrence. 16 persistent WGs, one 16-row batch tile each.
// h double-buffered in LDS (fp32 + bf16). Weights streamed bf16 from L2.
// 8 waves/WG; each wave owns 4 hidden-column 16-tiles and computes the six
// (gi,gh) x (r,z,n) accumulators via v_wmma_f32_16x16x32_bf16.
// K-loops are software-pipelined: fragment loads for k+1 issue before the
// WMMAs of k so L2 latency hides under the matrix ops.
// ---------------------------------------------------------------------------
__launch_bounds__(256, 1)
__global__ void gru_kernel(const float* __restrict__ x,
                           const float* __restrict__ h0,
                           const unsigned short* __restrict__ wih_u,
                           const unsigned short* __restrict__ whh_u,
                           float* __restrict__ hfinal) {
  extern __shared__ __align__(16) char smem[];
  float*  hf = (float*)smem;                                   // 2*16*512 f32  = 64 KB
  __bf16* hb = (__bf16*)(smem + 2 * BT * HIDDEN * 4);          // 2*16*512 bf16 = 32 KB
  __bf16* xb = (__bf16*)(smem + 2 * BT * HIDDEN * 4 + 2 * BT * HIDDEN * 2); // 8 KB

  const __bf16* wih = (const __bf16*)wih_u;
  const __bf16* whh = (const __bf16*)whh_u;

  const int bg   = blockIdx.x;
  const int tid  = threadIdx.x;
  const int wid  = tid >> 5;
  const int lane = tid & 31;
  const int m    = lane & 15;
  const int half = lane >> 4;

  // load initial hidden state for this batch tile
  for (int i = tid; i < BT * HIDDEN; i += 256) {
    float v = h0[(size_t)(bg * BT + (i >> 9)) * HIDDEN + (i & 511)];
    hf[i] = v;
    hb[i] = (__bf16)v;
  }
  __syncthreads();

  int cur = 0;
  for (int t = 0; t < SEQ; ++t) {
    // stage x_t tile into LDS as bf16
    const float* xt = x + ((size_t)t * BATCH + bg * BT) * INPUT_SZ;
    for (int i = tid; i < BT * INPUT_SZ; i += 256) xb[i] = (__bf16)xt[i];
    __syncthreads();

    const float*  hfc = hf + cur * BT * HIDDEN;
    const __bf16* hbc = hb + cur * BT * HIDDEN;
    float*  hfn = hf + (cur ^ 1) * BT * HIDDEN;
    __bf16* hbn = hb + (cur ^ 1) * BT * HIDDEN;

    for (int tile = 0; tile < 4; ++tile) {
      const int c0 = (wid * 4 + tile) * 16;
      v8f gh_r = {}, gh_z = {}, gh_n = {}, gi_r = {}, gi_z = {}, gi_n = {};

      // -- gh = h @ w_hh^T, K = 512, software-pipelined --
      {
        v16bf a  = load_a_frag(hbc, HIDDEN, m, half, 0);
        v16bf br = load_b_frag(whh, HIDDEN, c0,              0, lane);
        v16bf bz = load_b_frag(whh, HIDDEN, c0 + HIDDEN,     0, lane);
        v16bf bn = load_b_frag(whh, HIDDEN, c0 + 2 * HIDDEN, 0, lane);
#pragma unroll
        for (int k0 = 0; k0 < HIDDEN; k0 += 32) {
          v16bf ac = a, brc = br, bzc = bz, bnc = bn;
          const int kn = k0 + 32;
          if (kn < HIDDEN) {
            __builtin_prefetch(whh + (size_t)(c0 + m) * HIDDEN + kn + 32, 0, 3);
            a  = load_a_frag(hbc, HIDDEN, m, half, kn);
            br = load_b_frag(whh, HIDDEN, c0,              kn, lane);
            bz = load_b_frag(whh, HIDDEN, c0 + HIDDEN,     kn, lane);
            bn = load_b_frag(whh, HIDDEN, c0 + 2 * HIDDEN, kn, lane);
          }
          gh_r = wmma_bf16(ac, brc, gh_r);
          gh_z = wmma_bf16(ac, bzc, gh_z);
          gh_n = wmma_bf16(ac, bnc, gh_n);
        }
      }
      // -- gi = x_t @ w_ih^T, K = 256, software-pipelined --
      {
        v16bf a  = load_a_frag(xb, INPUT_SZ, m, half, 0);
        v16bf br = load_b_frag(wih, INPUT_SZ, c0,              0, lane);
        v16bf bz = load_b_frag(wih, INPUT_SZ, c0 + HIDDEN,     0, lane);
        v16bf bn = load_b_frag(wih, INPUT_SZ, c0 + 2 * HIDDEN, 0, lane);
#pragma unroll
        for (int k0 = 0; k0 < INPUT_SZ; k0 += 32) {
          v16bf ac = a, brc = br, bzc = bz, bnc = bn;
          const int kn = k0 + 32;
          if (kn < INPUT_SZ) {
            a  = load_a_frag(xb, INPUT_SZ, m, half, kn);
            br = load_b_frag(wih, INPUT_SZ, c0,              kn, lane);
            bz = load_b_frag(wih, INPUT_SZ, c0 + HIDDEN,     kn, lane);
            bn = load_b_frag(wih, INPUT_SZ, c0 + 2 * HIDDEN, kn, lane);
          }
          gi_r = wmma_bf16(ac, brc, gi_r);
          gi_z = wmma_bf16(ac, bzc, gi_z);
          gi_n = wmma_bf16(ac, bnc, gi_n);
        }
      }

      // fuse gates + GRU update; write into ping-pong buffer
#pragma unroll
      for (int r = 0; r < 8; ++r) {
        const int row = r + half * 8;
        const int col = c0 + m;
        float hold = hfc[row * HIDDEN + col];
        float rg = sigmoidf_(gi_r[r] + gh_r[r]);
        float zg = sigmoidf_(gi_z[r] + gh_z[r]);
        float ng = tanhf(gi_n[r] + rg * gh_n[r]);
        float hn = (1.0f - zg) * ng + zg * hold;
        hfn[row * HIDDEN + col] = hn;
        hbn[row * HIDDEN + col] = (__bf16)hn;
      }
    }
    __syncthreads();
    cur ^= 1;
  }

  const float* hres = hf + cur * BT * HIDDEN;
  for (int i = tid; i < BT * HIDDEN; i += 256)
    hfinal[(size_t)(bg * BT + (i >> 9)) * HIDDEN + (i & 511)] = hres[i];
}

// ---------------------------------------------------------------------------
// Kernel 2: alpha = softmax(h @ w_alpha^T), sigma logits -> inv_sigma + logdet,
//           mu = alpha @ external. WMMA for both GEMMs vs h; softmax in LDS.
// ---------------------------------------------------------------------------
__launch_bounds__(256, 1)
__global__ void attn_kernel(const float* __restrict__ hfinal,
                            const float* __restrict__ external,
                            const unsigned short* __restrict__ walpha_u,
                            const unsigned short* __restrict__ wsigma_u,
                            float* __restrict__ mu,
                            float* __restrict__ invsig,
                            float* __restrict__ logdet) {
  extern __shared__ __align__(16) char smem[];
  float*  logits  = (float*)smem;                                  // 16*2048 f32 = 128 KB
  __bf16* hb      = (__bf16*)(smem + BT * HIST * 4);               // 16*512 bf16 = 16 KB
  float*  red     = (float*)(smem + BT * HIST * 4 + BT * HIDDEN * 2); // 16*16
  float*  rowstat = red + 256;                                     // 16
  float*  ldet    = rowstat + 16;                                  // 16

  const __bf16* walpha = (const __bf16*)walpha_u;
  const __bf16* wsigma = (const __bf16*)wsigma_u;

  const int bg   = blockIdx.x;
  const int tid  = threadIdx.x;
  const int wid  = tid >> 5;
  const int lane = tid & 31;
  const int m    = lane & 15;
  const int half = lane >> 4;

  for (int i = tid; i < BT * HIDDEN; i += 256)
    hb[i] = (__bf16)hfinal[(size_t)(bg * BT + (i >> 9)) * HIDDEN + (i & 511)];
  if (tid < BT) ldet[tid] = 0.0f;
  __syncthreads();

  // sigma logits [16 x 256]; sigma = exp(v) so log|sigma| = v, 1/sigma = exp(-v)
#pragma unroll
  for (int tt = 0; tt < 2; ++tt) {
    const int j0 = (wid * 2 + tt) * 16;
    v8f acc = {};
#pragma unroll
    for (int k0 = 0; k0 < HIDDEN; k0 += 32) {
      v16bf a = load_a_frag(hb, HIDDEN, m, half, k0);
      acc = wmma_bf16(a, load_b_frag(wsigma, HIDDEN, j0, k0, lane), acc);
    }
#pragma unroll
    for (int r = 0; r < 8; ++r) {
      const int row = r + half * 8;
      float v = acc[r];
      invsig[(size_t)(bg * BT + row) * INPUT_SZ + j0 + m] = __expf(-v);
      atomicAdd(&ldet[row], v);
    }
  }

  // alpha logits [16 x 2048], software-pipelined over K
  for (int tt = 0; tt < 16; ++tt) {
    const int j0 = wid * 256 + tt * 16;
    v8f acc = {};
    v16bf a = load_a_frag(hb, HIDDEN, m, half, 0);
    v16bf b = load_b_frag(walpha, HIDDEN, j0, 0, lane);
#pragma unroll
    for (int k0 = 0; k0 < HIDDEN; k0 += 32) {
      v16bf ac = a, bc = b;
      const int kn = k0 + 32;
      if (kn < HIDDEN) {
        a = load_a_frag(hb, HIDDEN, m, half, kn);
        b = load_b_frag(walpha, HIDDEN, j0, kn, lane);
      }
      acc = wmma_bf16(ac, bc, acc);
    }
#pragma unroll
    for (int r = 0; r < 8; ++r)
      logits[(r + half * 8) * HIST + j0 + m] = acc[r];
  }
  __syncthreads();

  // row softmax: 16 threads per row
  const int srow = tid >> 4;
  const int sc   = tid & 15;
  float mx = -3.4e38f;
  for (int k = sc; k < HIST; k += 16) mx = fmaxf(mx, logits[srow * HIST + k]);
  red[srow * 16 + sc] = mx;
  __syncthreads();
  if (sc == 0) {
    float v = red[srow * 16];
    for (int i = 1; i < 16; ++i) v = fmaxf(v, red[srow * 16 + i]);
    rowstat[srow] = v;
  }
  __syncthreads();
  const float rmax = rowstat[srow];
  float sum = 0.0f;
  for (int k = sc; k < HIST; k += 16) {
    float e = __expf(logits[srow * HIST + k] - rmax);
    logits[srow * HIST + k] = e;
    sum += e;
  }
  red[srow * 16 + sc] = sum;
  __syncthreads();
  if (sc == 0) {
    float v = 0.0f;
    for (int i = 0; i < 16; ++i) v += red[srow * 16 + i];
    rowstat[srow] = v;
  }
  __syncthreads();
  const float rinv = 1.0f / rowstat[srow];
  for (int k = sc; k < HIST; k += 16) logits[srow * HIST + k] *= rinv;
  __syncthreads();

  if (tid < BT) logdet[bg * BT + tid] = ldet[tid];

  // mu = alpha @ external : [16 x 256]  (tiny; VALU, external streams from L2)
  const int mrow = tid & 15;
  const int d0   = (tid >> 4) * 16;
  float accm[16];
#pragma unroll
  for (int j = 0; j < 16; ++j) accm[j] = 0.0f;
  for (int k = 0; k < HIST; ++k) {
    float a = logits[mrow * HIST + k];
    const float* er = external + (size_t)k * INPUT_SZ + d0;
#pragma unroll
    for (int j = 0; j < 16; ++j) accm[j] += a * er[j];
  }
#pragma unroll
  for (int j = 0; j < 16; ++j)
    mu[(size_t)(bg * BT + mrow) * INPUT_SZ + d0 + j] = accm[j];
}

// ---------------------------------------------------------------------------
// Kernel 3: streaming Gaussian log-prob. One wave32 per (s,b); 8 elems/lane,
// shuffle-tree reduction. Bandwidth-bound re-read of x (~134 MB @ 23.3 TB/s).
// ---------------------------------------------------------------------------
__launch_bounds__(256)
__global__ void logprob_kernel(const float* __restrict__ x,
                               const float* __restrict__ mu,
                               const float* __restrict__ invsig,
                               const float* __restrict__ logdet,
                               float* __restrict__ out) {
  const int wid  = threadIdx.x >> 5;
  const int lane = threadIdx.x & 31;
  const int idx  = blockIdx.x * 8 + wid;       // [0, SEQ*BATCH)
  const int b    = idx & (BATCH - 1);
  const int s    = idx >> 8;
  const float* xp = x + ((size_t)s * BATCH + b) * INPUT_SZ + lane * 8;
  const float* mp = mu + (size_t)b * INPUT_SZ + lane * 8;
  const float* ip = invsig + (size_t)b * INPUT_SZ + lane * 8;
  float sum = 0.0f;
#pragma unroll
  for (int j = 0; j < 8; ++j) {
    float d = (xp[j] - mp[j]) * ip[j];
    sum += d * d;
  }
#pragma unroll
  for (int off = 16; off >= 1; off >>= 1) sum += __shfl_xor(sum, off, 32);
  if (lane == 0) {
    const float log2pi = 1.83787706640934548356f;
    out[(size_t)s * BATCH + b] =
        -0.5f * (sum + (float)INPUT_SZ * log2pi) - logdet[b];
  }
}

// ---------------------------------------------------------------------------
extern "C" void kernel_launch(void* const* d_in, const int* in_sizes, int n_in,
                              void* d_out, int out_size, void* d_ws, size_t ws_size,
                              hipStream_t stream) {
  const float* x        = (const float*)d_in[0];
  const float* hidden   = (const float*)d_in[1];
  const float* external = (const float*)d_in[2];
  const float* w_ih     = (const float*)d_in[3];
  const float* w_hh     = (const float*)d_in[4];
  const float* w_alpha  = (const float*)d_in[5];
  const float* w_sigma  = (const float*)d_in[6];
  float* out = (float*)d_out;

  char* ws = (char*)d_ws;
  size_t o = 0;
  unsigned short* wih_b    = (unsigned short*)(ws + o); o += (size_t)GATES * INPUT_SZ * 2;
  unsigned short* whh_b    = (unsigned short*)(ws + o); o += (size_t)GATES * HIDDEN * 2;
  unsigned short* walpha_b = (unsigned short*)(ws + o); o += (size_t)HIST * HIDDEN * 2;
  unsigned short* wsigma_b = (unsigned short*)(ws + o); o += (size_t)INPUT_SZ * HIDDEN * 2;
  float* hfinal = (float*)(ws + o); o += (size_t)BATCH * HIDDEN * 4;
  float* mu     = (float*)(ws + o); o += (size_t)BATCH * INPUT_SZ * 4;
  float* invs   = (float*)(ws + o); o += (size_t)BATCH * INPUT_SZ * 4;
  float* logdet = (float*)(ws + o); o += (size_t)BATCH * 4;

  cvt_bf16_kernel<<<256, 256, 0, stream>>>(w_ih,    wih_b,    GATES * INPUT_SZ);
  cvt_bf16_kernel<<<256, 256, 0, stream>>>(w_hh,    whh_b,    GATES * HIDDEN);
  cvt_bf16_kernel<<<256, 256, 0, stream>>>(w_alpha, walpha_b, HIST * HIDDEN);
  cvt_bf16_kernel<<<256, 256, 0, stream>>>(w_sigma, wsigma_b, INPUT_SZ * HIDDEN);

  const size_t smem1 = (size_t)2 * BT * HIDDEN * 4   // hf ping-pong
                     + (size_t)2 * BT * HIDDEN * 2   // hb ping-pong
                     + (size_t)BT * INPUT_SZ * 2;    // xb
  gru_kernel<<<BATCH / BT, 256, smem1, stream>>>(x, hidden, wih_b, whh_b, hfinal);

  const size_t smem2 = (size_t)BT * HIST * 4         // logits/alpha
                     + (size_t)BT * HIDDEN * 2       // hb
                     + (size_t)(256 + 16 + 16) * 4;  // reductions
  attn_kernel<<<BATCH / BT, 256, smem2, stream>>>(hfinal, external, walpha_b,
                                                  wsigma_b, mu, invs, logdet);

  logprob_kernel<<<(SEQ * BATCH) / 8, 256, 0, stream>>>(x, mu, invs, logdet, out);
}
